// IndividualAutoEncoder_10995116278476
// MI455X (gfx1250) — matrix operand order repaired and checked by reading, not compiled, not run on hardware
//
#include <hip/hip_runtime.h>
#include <stdint.h>

typedef __attribute__((ext_vector_type(16))) __bf16        v16bf;
typedef __attribute__((ext_vector_type(8)))  float         v8f;
typedef __attribute__((ext_vector_type(8)))  unsigned int  v8u;

#define NB 512   // batch
#define NS 2048  // seq len
#define NV 29    // vocab / input size
#define NH 128   // hidden
#define VP 32    // padded V (K must be 32 for wmma)

// Round-to-nearest-even f32->bf16: used only in one-time weight preload.
__device__ __forceinline__ unsigned short f2bf(float f) {
  union { float f; unsigned int u; } c; c.f = f;
  unsigned int u = c.u + 0x7FFFu + ((c.u >> 16) & 1u);
  return (unsigned short)(u >> 16);
}

// Truncating f32->bf16: free in the hot loop (ds_store_b16_d16_hi takes bits[31:16]).
__device__ __forceinline__ unsigned short bf_trunc(float f) {
  return (unsigned short)(__builtin_bit_cast(unsigned int, f) >> 16);
}

// Native CDNA5 transcendental tanh (V_TANH_F32) instead of branchy libm tanhf.
__device__ __forceinline__ float htanh(float v) {
#if __has_builtin(__builtin_amdgcn_tanhf)
  return __builtin_amdgcn_tanhf(v);
#else
  float r;
  // TRANS op: needs 1 independent op / V_NOP before result use (ISA §7.4).
  asm("v_tanh_f32 %0, %1\n\tv_nop" : "=v"(r) : "v"(v));
  return r;
#endif
}

__device__ __forceinline__ v16bf bits_to_bf(v8u a) {
  return __builtin_bit_cast(v16bf, a);
}

// Build a 16x32 bf16 WMMA A fragment from a row-major LDS tile.
// Layout used here (contiguous k, same as B): lane m=L&15, hi=L>>4;
// element j -> K = j + 16*hi  => two contiguous 16B chunks.
__device__ __forceinline__ v16bf load_a_frag(const unsigned short* base, int rowStride,
                                             int kBase, int lane) {
  const int m  = lane & 15;
  const int hi = lane >> 4;
  const unsigned short* p = base + m * rowStride + kBase + 16 * hi;
  uint4 c0 = *(const uint4*)(p);        // K = kBase+16*hi  .. +7
  uint4 c1 = *(const uint4*)(p + 8);    // K = kBase+16*hi+8.. +7
  v8u a;
  a[0] = c0.x; a[1] = c0.y; a[2] = c0.z; a[3] = c0.w;
  a[4] = c1.x; a[5] = c1.y; a[6] = c1.z; a[7] = c1.w;
  return bits_to_bf(a);
}

__global__ __launch_bounds__(256, 1)
void rnn_autoenc_kernel(const float* __restrict__ x,
                        const float* __restrict__ We_ih, const float* __restrict__ We_hh,
                        const float* __restrict__ be_ih, const float* __restrict__ be_hh,
                        const float* __restrict__ Wd_ih, const float* __restrict__ Wd_hh,
                        const float* __restrict__ bd_ih, const float* __restrict__ bd_hh,
                        float* __restrict__ out) {
  __shared__ __align__(16) unsigned short sh_h[2][16 * NH];   // encoder hidden, double-buffered
  __shared__ __align__(16) unsigned short sh_x[2][16 * VP];   // x tile, double-buffered
  __shared__ __align__(16) unsigned short sh_hd[2][16 * VP];  // decoder hidden, double-buffered

  const int tid  = threadIdx.x;
  const int wave = tid >> 5;
  const int lane = tid & 31;
  const int nl   = lane & 15;
  const int hi   = lane >> 4;
  const int b0   = blockIdx.x * 16;

  // ---------- preload encoder B fragments (wave owns N-tile 'wave') ----------
  // B layout used here (contiguous k, same as A): N = nl, element j -> K = j + 16*hi.
  // B[k][n] = W[n][k]  (we compute h @ W^T), so fragment = 16 contiguous f32 of W row n.
  const int nG = wave * 16 + nl;  // global hidden column 0..127
  v16bf Bhh[4];
#pragma unroll
  for (int kc = 0; kc < 4; ++kc) {
    const float* wr = We_hh + nG * NH + kc * 32 + 16 * hi;
    v8u tb;
#pragma unroll
    for (int p = 0; p < 8; ++p)
      tb[p] = (unsigned int)f2bf(wr[2 * p]) | ((unsigned int)f2bf(wr[2 * p + 1]) << 16);
    Bhh[kc] = bits_to_bf(tb);
  }
  v16bf Bih;
  {
    const float* wr = We_ih + nG * NV;
    v8u tb;
#pragma unroll
    for (int p = 0; p < 8; ++p) {
      int k0 = 16 * hi + 2 * p, k1 = k0 + 1;
      float f0 = (k0 < NV) ? wr[k0] : 0.f;
      float f1 = (k1 < NV) ? wr[k1] : 0.f;
      tb[p] = (unsigned int)f2bf(f0) | ((unsigned int)f2bf(f1) << 16);
    }
    Bih = bits_to_bf(tb);
  }
  const float ebias = be_ih[nG] + be_hh[nG];  // same for all 8 C rows of this lane

  // x tile prefetch addressing: 16x32 elems, 2 per thread; unconditional clamped load + mask
  const int e0 = tid, e1 = tid + 256;
  const int r0 = e0 >> 5, c0 = e0 & 31;
  const int r1 = e1 >> 5, c1 = e1 & 31;
  const int c0c = (c0 < NV) ? c0 : (NV - 1);
  const int c1c = (c1 < NV) ? c1 : (NV - 1);
  const float m0 = (c0 < NV) ? 1.f : 0.f;
  const float m1 = (c1 < NV) ? 1.f : 0.f;
  const float* xp0 = x + (size_t)(b0 + r0) * NS * NV + c0c;
  const float* xp1 = x + (size_t)(b0 + r1) * NS * NV + c1c;

  // zero initial hidden state (buffer 0) and fill x tile for t=0 (buffer 0)
  for (int e = tid; e < 16 * NH; e += 256) sh_h[0][e] = 0;
  sh_x[0][e0] = bf_trunc(xp0[0] * m0);
  sh_x[0][e1] = bf_trunc(xp1[0] * m1);
  __syncthreads();

  // ---------- encoder recurrence: h = tanh(x_t@We_ih^T + h@We_hh^T + b) ----------
  int buf = 0;
  for (int t = 0; t < NS; ++t) {
    // issue next x-tile loads early; latency hides under the WMMA chain
    const int tn = (t + 1 < NS) ? (t + 1) : (NS - 1);
    const float nf0 = xp0[(size_t)tn * NV] * m0;
    const float nf1 = xp1[(size_t)tn * NV] * m1;

    // all 10 ds_load_b128 up front, then the serially-dependent WMMA chain
    v16bf Ax  = load_a_frag(sh_x[buf], VP, 0, lane);
    v16bf Ah0 = load_a_frag(sh_h[buf], NH, 0,  lane);
    v16bf Ah1 = load_a_frag(sh_h[buf], NH, 32, lane);
    v16bf Ah2 = load_a_frag(sh_h[buf], NH, 64, lane);
    v16bf Ah3 = load_a_frag(sh_h[buf], NH, 96, lane);

    v8f c;
#pragma unroll
    for (int r = 0; r < 8; ++r) c[r] = ebias;
    c = __builtin_amdgcn_wmma_f32_16x16x32_bf16(false, Ax,  false, Bih,    (short)0, c, false, false);
    c = __builtin_amdgcn_wmma_f32_16x16x32_bf16(false, Ah0, false, Bhh[0], (short)0, c, false, false);
    c = __builtin_amdgcn_wmma_f32_16x16x32_bf16(false, Ah1, false, Bhh[1], (short)0, c, false, false);
    c = __builtin_amdgcn_wmma_f32_16x16x32_bf16(false, Ah2, false, Bhh[2], (short)0, c, false, false);
    c = __builtin_amdgcn_wmma_f32_16x16x32_bf16(false, Ah3, false, Bhh[3], (short)0, c, false, false);

#pragma unroll
    for (int r = 0; r < 8; ++r) {           // D layout: VGPR r -> row (r + 8*hi), col nG
      float hv = htanh(c[r]);
      // truncating bf16: ds_store_b16_d16_hi consumes the tanh register directly
      sh_h[buf ^ 1][(r + 8 * hi) * NH + nG] = bf_trunc(hv);
    }
    sh_x[buf ^ 1][e0] = bf_trunc(nf0);
    sh_x[buf ^ 1][e1] = bf_trunc(nf1);
    __syncthreads();                         // single barrier per step (double-buffered)
    buf ^= 1;
  }
  const unsigned short* enc = sh_h[buf];     // encoded (16 x 128, bf16)

  // ---------- decoder setup: dec_in = encoded@Wd_ih^T + bd (waves 0,1 only) ----------
  v16bf Bdh = {};
  v8f dec_in = {};
  const int nD = wave * 16 + nl;  // decoder output column for waves 0/1
  if (wave < 2) {
    float dbias = (nD < NV) ? (bd_ih[nD] + bd_hh[nD]) : 0.f;
    v8f c;
#pragma unroll
    for (int r = 0; r < 8; ++r) c[r] = dbias;
#pragma unroll
    for (int kc = 0; kc < 4; ++kc) {
      v8u tb;
#pragma unroll
      for (int p = 0; p < 8; ++p) {
        int k0 = kc * 32 + 16 * hi + 2 * p;
        float f0 = (nD < NV) ? Wd_ih[nD * NH + k0]     : 0.f;
        float f1 = (nD < NV) ? Wd_ih[nD * NH + k0 + 1] : 0.f;
        tb[p] = (unsigned int)f2bf(f0) | ((unsigned int)f2bf(f1) << 16);
      }
      v16bf Bd = bits_to_bf(tb);
      v16bf Ae = load_a_frag(enc, NH, kc * 32, lane);
      c = __builtin_amdgcn_wmma_f32_16x16x32_bf16(false, Ae, false, Bd, (short)0, c, false, false);
    }
    dec_in = c;
    // Wd_hh (29x29) fragment, zero-padded to 32x32: pad rows kill junk in padded A cols
    v8u tb;
#pragma unroll
    for (int p = 0; p < 8; ++p) {
      int k0 = 16 * hi + 2 * p, k1 = k0 + 1;
      float f0 = (nD < NV && k0 < NV) ? Wd_hh[nD * NV + k0] : 0.f;
      float f1 = (nD < NV && k1 < NV) ? Wd_hh[nD * NV + k1] : 0.f;
      tb[p] = (unsigned int)f2bf(f0) | ((unsigned int)f2bf(f1) << 16);
    }
    Bdh = bits_to_bf(tb);
  }
  for (int e = tid; e < 16 * VP; e += 256) sh_hd[0][e] = 0;
  __syncthreads();

  // ---------- decoder recurrence: h = tanh(dec_in + h@Wd_hh^T), emit every step ----------
  int db = 0;
  for (int t = 0; t < NS; ++t) {
    if (wave < 2) {
      v16bf Ad = load_a_frag(sh_hd[db], VP, 0, lane);
      v8f c = __builtin_amdgcn_wmma_f32_16x16x32_bf16(false, Ad, false, Bdh, (short)0,
                                                      dec_in, false, false);
#pragma unroll
      for (int r = 0; r < 8; ++r) {
        float hv = htanh(c[r]);
        int m = r + 8 * hi;
        sh_hd[db ^ 1][m * VP + nD] = bf_trunc(hv);  // other buffer: no WAR hazard, one barrier
        if (nD < NV)
          out[((size_t)(b0 + m) * NS + t) * NV + nD] = hv;
      }
    }
    __syncthreads();
    db ^= 1;
  }
}

extern "C" void kernel_launch(void* const* d_in, const int* in_sizes, int n_in,
                              void* d_out, int out_size, void* d_ws, size_t ws_size,
                              hipStream_t stream) {
  const float* x     = (const float*)d_in[0];
  const float* We_ih = (const float*)d_in[1];
  const float* We_hh = (const float*)d_in[2];
  const float* be_ih = (const float*)d_in[3];
  const float* be_hh = (const float*)d_in[4];
  const float* Wd_ih = (const float*)d_in[5];
  const float* Wd_hh = (const float*)d_in[6];
  const float* bd_ih = (const float*)d_in[7];
  const float* bd_hh = (const float*)d_in[8];
  float* out = (float*)d_out;
  rnn_autoenc_kernel<<<dim3(NB / 16), dim3(256), 0, stream>>>(
      x, We_ih, We_hh, be_ih, be_hh, Wd_ih, Wd_hh, bd_ih, bd_hh, out);
}
